// GAT_residue_56581899157773
// MI455X (gfx1250) — matrix operand order repaired and checked by reading, not compiled, not run on hardware
//
#include <hip/hip_runtime.h>

typedef __attribute__((ext_vector_type(16))) _Float16 v16h;
typedef __attribute__((ext_vector_type(8)))  float    v8f;
typedef __attribute__((ext_vector_type(4)))  unsigned tdm_u4;
typedef __attribute__((ext_vector_type(8)))  int      tdm_i8;
typedef __attribute__((ext_vector_type(4)))  int      tdm_i4;

#if defined(__has_builtin)
#if __has_builtin(__builtin_amdgcn_tensor_load_to_lds) && __has_builtin(__builtin_amdgcn_s_wait_tensorcnt)
#define USE_TDM 1
#endif
#endif

#define N_NODES 20000
#define IN_DIM  512
#define HIDTOT  512     // HEADS * HID
#define HEADS   4
#define OUT_DIM 64
#define E_RAW   320000
#define E_TOT   340000  // + self loops
#define NEG_SLOPE 0.2f

// ---------------- order-preserving float <-> uint key for atomic max ----------------
__device__ __forceinline__ unsigned fkey(float f) {
    unsigned u = __float_as_uint(f);
    return (u & 0x80000000u) ? ~u : (u | 0x80000000u);
}
__device__ __forceinline__ float funkey(unsigned k) {
    unsigned u = (k & 0x80000000u) ? (k & 0x7FFFFFFFu) : ~k;
    return __uint_as_float(u);
}

__device__ __forceinline__ void edge_nodes(const int* __restrict__ ei, int e, int& src, int& dst) {
    if (e < E_RAW) { src = ei[e]; dst = ei[E_RAW + e]; }
    else           { src = e - E_RAW; dst = src; }
}

// ---------------- f16 staging (removes cvt + guards from GEMM hot loop) ----------------
__launch_bounds__(256)
__global__ void gat_cvt_f16(const float* __restrict__ in, _Float16* __restrict__ out, int n)
{
    int i = blockIdx.x * blockDim.x + threadIdx.x;
    if (i < n) out[i] = (_Float16)in[i];
}

__launch_bounds__(256)
__global__ void gat_relu_bias_f16(const float* __restrict__ in, const float* __restrict__ b,
                                  _Float16* __restrict__ out, int total, int C)
{
    int i = blockIdx.x * blockDim.x + threadIdx.x;
    if (i < total) {
        float v = in[i] + b[i % C];
        out[i] = (_Float16)(v > 0.f ? v : 0.f);
    }
}

// ---------------- f16 GEMM with f32 accumulate: C[M,N] = A[M,K] @ B[K,N] ----------------
// Block = 128 threads (4 waves), block tile 64x64, wave tile 32x32 (2x2 WMMA), K-step 32.
// A tile streamed into LDS by the Tensor Data Mover (async, OOB zero-fill, HW padding);
// B tile transposed manually into LDS (loads hoisted, no divergence).
#define BM 64
#define BN 64
#define BK 32

__launch_bounds__(128)
__global__ void gat_gemm_wmma(const _Float16* __restrict__ A, const _Float16* __restrict__ B,
                              float* __restrict__ C, int M, int N, int K)
{
    __shared__ _Float16 As[BM][BK + 2];   // stride 34 halfs == TDM pad 1 DWORD / 16 DWORDs
    __shared__ _Float16 Bs[BN][BK + 2];   // transposed: [col][k]

    const int tid  = threadIdx.x;
    const int lane = tid & 31;
    const int wave = tid >> 5;
    const int g    = lane >> 4;
    const int l16  = lane & 15;

    const int blockRow = blockIdx.y * BM;
    const int blockCol = blockIdx.x * BN;
    const int wr = (wave >> 1) * 32;
    const int wc = (wave & 1) * 32;

    v8f acc[2][2] = {};

#ifdef USE_TDM
    // D# group 1: constant across K steps. data_size=2B, pad_enable, pad_interval=16 DWORDs,
    // pad_amount=1 DWORD; tensor_dim0=K (X always in-bounds), tensor_dim1 = rows remaining
    // from tile start (zero-fill beyond M); tile = 32 x 64; stride0 = K elements.
    const unsigned dim1 = (unsigned)(M - blockRow);
    tdm_i8 g1;
    g1[0] = (int)((1u << 16) | (1u << 20) | (3u << 22));
    g1[1] = (int)(((unsigned)K & 0xFFFFu) << 16);
    g1[2] = (int)(((unsigned)K >> 16) | ((dim1 & 0xFFFFu) << 16));
    g1[3] = (int)((dim1 >> 16) | ((unsigned)BK << 16));
    g1[4] = BM;              // tile_dim1 = 64, tile_dim2 = 0
    g1[5] = K;               // tensor_dim0_stride low 32 (elements)
    g1[6] = 0;
    g1[7] = 0;
    const tdm_i4 gz4 = {0, 0, 0, 0};
    const tdm_i8 gz8 = {0, 0, 0, 0, 0, 0, 0, 0};
    const unsigned ldsA = (unsigned)(uintptr_t)&As[0][0];
    const unsigned long long gbase =
        (unsigned long long)(uintptr_t)A + 2ull * ((unsigned long long)blockRow * (unsigned)K);
#endif

    for (int kb = 0; kb < K; kb += BK) {
#ifdef USE_TDM
        if (wave == 0) {
            unsigned long long gaddr = gbase + 2ull * (unsigned)kb;
            tdm_u4 g0;
            g0[0] = 1u;                                   // count=1
            g0[1] = ldsA;                                 // LDS byte address
            g0[2] = (unsigned)(gaddr & 0xFFFFFFFFull);
            g0[3] = (unsigned)((gaddr >> 32) & 0x1FFFFFFull) | (2u << 30);  // type=2
            __builtin_amdgcn_tensor_load_to_lds(g0, g1, gz4, gz4, gz8, 0);
        }
#else
        {   // manual A tile: clamp addresses (always in-bounds), mask data; loads hoisted
            int r = tid >> 2;                 // 0..31
            int c = (tid & 3) * 8;            // 0..24
            union { uint4 q; unsigned u[4]; } q0, q1;
            int r0 = blockRow + r, r1 = blockRow + r + 32;
            int a0 = r0 < M ? r0 : M - 1, a1 = r1 < M ? r1 : M - 1;
            q0.q = *(const uint4*)(A + (size_t)a0 * K + kb + c);
            q1.q = *(const uint4*)(A + (size_t)a1 * K + kb + c);
            if (r0 >= M) q0.q = make_uint4(0u, 0u, 0u, 0u);
            if (r1 >= M) q1.q = make_uint4(0u, 0u, 0u, 0u);
            #pragma unroll
            for (int i = 0; i < 4; ++i) {
                *(unsigned*)&As[r][c + 2 * i]      = q0.u[i];
                *(unsigned*)&As[r + 32][c + 2 * i] = q1.u[i];
            }
        }
#endif
        {   // B tile 32x64 f16 -> transposed LDS [n][k]; both loads issued before stores
            int k = tid >> 3;                 // 0..15
            int c = (tid & 7) * 8;            // 0..56
            union { uint4 q; _Float16 h[8]; } q0, q1;
            q0.q = *(const uint4*)(B + (size_t)(kb + k) * N + blockCol + c);
            q1.q = *(const uint4*)(B + (size_t)(kb + k + 16) * N + blockCol + c);
            #pragma unroll
            for (int i = 0; i < 8; ++i) {
                Bs[c + i][k]      = q0.h[i];
                Bs[c + i][k + 16] = q1.h[i];
            }
        }
#ifdef USE_TDM
        __builtin_amdgcn_s_wait_tensorcnt(0);  // no-op for waves 1..3 (TENSORcnt==0)
#endif
        __syncthreads();

        #pragma unroll
        for (int i = 0; i < 2; ++i) {
            union { v16h v; unsigned u[8]; } af;
            int arow = wr + i * 16 + l16;
            #pragma unroll
            for (int vv = 0; vv < 8; ++vv) {
                int kk = (vv < 4) ? (2 * vv + 8 * g) : (16 + 2 * (vv - 4) + 8 * g);
                af.u[vv] = *(const unsigned*)&As[arow][kk];
            }
            #pragma unroll
            for (int j = 0; j < 2; ++j) {
                union { v16h v; unsigned u[8]; } bf;
                int bcol = wc + j * 16 + l16;
                #pragma unroll
                for (int vv = 0; vv < 8; ++vv) {
                    int kk = (vv < 4) ? (2 * vv + 8 * g) : (16 + 2 * (vv - 4) + 8 * g);
                    bf.u[vv] = *(const unsigned*)&Bs[bcol][kk];
                }
                acc[i][j] = __builtin_amdgcn_wmma_f32_16x16x32_f16(
                    false, af.v, false, bf.v, (short)0, acc[i][j], false, false);
            }
        }
        __syncthreads();
    }

    #pragma unroll
    for (int i = 0; i < 2; ++i)
        #pragma unroll
        for (int j = 0; j < 2; ++j)
            #pragma unroll
            for (int r = 0; r < 8; ++r) {
                int grow = blockRow + wr + i * 16 + r + 8 * g;
                int gcol = blockCol + wc + j * 16 + l16;
                if (grow < M) C[(size_t)grow * N + gcol] = acc[i][j][r];
            }
}

// ---------------- per-node attention coefficients ----------------
__launch_bounds__(128)
__global__ void gat_alpha_h4(const float* __restrict__ h, const float* __restrict__ a_s,
                             const float* __restrict__ a_d,
                             float* __restrict__ as_out, float* __restrict__ ad_out)
{
    int n = blockIdx.x, t = threadIdx.x;
    int head = t >> 5, lane = t & 31;
    float4 hv = *(const float4*)(h + (size_t)n * HIDTOT + t * 4);
    float4 sv = *(const float4*)(a_s + t * 4);
    float4 dv = *(const float4*)(a_d + t * 4);
    float ps = hv.x * sv.x + hv.y * sv.y + hv.z * sv.z + hv.w * sv.w;
    float pd = hv.x * dv.x + hv.y * dv.y + hv.z * dv.z + hv.w * dv.w;
    for (int o = 16; o > 0; o >>= 1) { ps += __shfl_down(ps, o, 32); pd += __shfl_down(pd, o, 32); }
    if (lane == 0) { as_out[n * HEADS + head] = ps; ad_out[n * HEADS + head] = pd; }
}

__launch_bounds__(64)
__global__ void gat_alpha_h1(const float* __restrict__ h, const float* __restrict__ a_s,
                             const float* __restrict__ a_d,
                             float* __restrict__ as_out, float* __restrict__ ad_out)
{
    __shared__ float sp[2], dp[2];
    int n = blockIdx.x, t = threadIdx.x, lane = t & 31, w = t >> 5;
    float hv = h[(size_t)n * OUT_DIM + t];
    float ps = hv * a_s[t], pd = hv * a_d[t];
    for (int o = 16; o > 0; o >>= 1) { ps += __shfl_down(ps, o, 32); pd += __shfl_down(pd, o, 32); }
    if (lane == 0) { sp[w] = ps; dp[w] = pd; }
    __syncthreads();
    if (t == 0) { as_out[n] = sp[0] + sp[1]; ad_out[n] = dp[0] + dp[1]; }
}

// ---------------- segment softmax over edges ----------------
__launch_bounds__(256)
__global__ void gat_edge_max(const int* __restrict__ ei, int H,
                             const float* __restrict__ as, const float* __restrict__ ad,
                             unsigned* __restrict__ emax)
{
    int idx = blockIdx.x * blockDim.x + threadIdx.x;
    if (idx >= E_TOT * H) return;
    int e = idx / H, h = idx % H;
    int src, dst; edge_nodes(ei, e, src, dst);
    float ev = as[src * H + h] + ad[dst * H + h];
    ev = ev > 0.f ? ev : NEG_SLOPE * ev;
    atomicMax(&emax[dst * H + h], fkey(ev));
}

__launch_bounds__(256)
__global__ void gat_edge_exp(const int* __restrict__ ei, int H,
                             const float* __restrict__ as, const float* __restrict__ ad,
                             const unsigned* __restrict__ emax,
                             float* __restrict__ eexp, float* __restrict__ denom)
{
    int idx = blockIdx.x * blockDim.x + threadIdx.x;
    if (idx >= E_TOT * H) return;
    int e = idx / H, h = idx % H;
    int src, dst; edge_nodes(ei, e, src, dst);
    float ev = as[src * H + h] + ad[dst * H + h];
    ev = ev > 0.f ? ev : NEG_SLOPE * ev;
    float w = __expf(ev - funkey(emax[dst * H + h]));
    eexp[idx] = w;                        // idx == e*H + h
    atomicAdd(&denom[dst * H + h], w);
}

// One wave per edge; lanes stride channels (coalesced); atomicAdd into L2-resident agg.
__launch_bounds__(256)
__global__ void gat_edge_agg(const int* __restrict__ ei, int H, int C,
                             const float* __restrict__ hf, const float* __restrict__ eexp,
                             const float* __restrict__ denom, float* __restrict__ agg)
{
    int w = (blockIdx.x * blockDim.x + threadIdx.x) >> 5;
    int lane = threadIdx.x & 31;
    if (w >= E_TOT) return;
    int src, dst; edge_nodes(ei, w, src, dst);
    int cph = C / H;
    const float* hrow = hf + (size_t)src * C;
    float* arow = agg + (size_t)dst * C;
    for (int c = lane; c < C; c += 32) {
        int head = c / cph;
        float alpha = eexp[w * H + head] / (denom[dst * H + head] + 1e-16f);
        atomicAdd(&arow[c], hrow[c] * alpha);
    }
}

__launch_bounds__(256)
__global__ void gat_bias_out(const float* __restrict__ a, const float* __restrict__ b,
                             float* __restrict__ out, int total, int C)
{
    int i = blockIdx.x * blockDim.x + threadIdx.x;
    if (i < total) out[i] = a[i] + b[i % C];
}

// ---------------- launch ----------------
extern "C" void kernel_launch(void* const* d_in, const int* in_sizes, int n_in,
                              void* d_out, int out_size, void* d_ws, size_t ws_size,
                              hipStream_t stream)
{
    (void)in_sizes; (void)n_in; (void)out_size; (void)ws_size;

    const float* x      = (const float*)d_in[0];
    const int*   ei     = (const int*)  d_in[1];
    const float* W1     = (const float*)d_in[2];
    const float* a_src1 = (const float*)d_in[3];
    const float* a_dst1 = (const float*)d_in[4];
    const float* b1     = (const float*)d_in[5];
    const float* W2     = (const float*)d_in[6];
    const float* a_src2 = (const float*)d_in[7];
    const float* a_dst2 = (const float*)d_in[8];
    const float* b2     = (const float*)d_in[9];
    float* out = (float*)d_out;
    char*  ws  = (char*)d_ws;

    // workspace layout (bytes); f16 staging buffers alias dead f32 regions
    float*     h1    = (float*)    (ws + 0);            // 40,960,000 (layer-1 features)
    float*     h2    = (float*)    (ws + 0);            //  5,120,000 (after h1 dies)
    _Float16*  ah2   = (_Float16*) (ws + 5120000);      // 20,480,000 (relu act, f16)
    _Float16*  W2h   = (_Float16*) (ws + 25600000);     //     65,536
    float*     agg1  = (float*)    (ws + 40960000);     // 40,960,000
    _Float16*  xh    = (_Float16*) (ws + 40960000);     // 20,480,000 (dies before agg1 memset)
    _Float16*  W1h   = (_Float16*) (ws + 61440000);     //    524,288
    float*     as1   = (float*)    (ws + 81920000);
    float*     ad1   = (float*)    (ws + 82240000);
    unsigned*  emax1 = (unsigned*) (ws + 82560000);
    float*     den1  = (float*)    (ws + 82880000);
    float*     eexp1 = (float*)    (ws + 83200000);     //  5,440,000
    float*     as2   = (float*)    (ws + 88640000);
    float*     ad2   = (float*)    (ws + 88720000);
    unsigned*  emax2 = (unsigned*) (ws + 88800000);
    float*     den2  = (float*)    (ws + 88880000);
    float*     eexp2 = (float*)    (ws + 88960000);     //  1,360,000
    float*     agg2  = (float*)    (ws + 90320000);     //  5,120,000

    // ---- layer 1 ----
    int nx = N_NODES * IN_DIM, nw1 = IN_DIM * HIDTOT;
    gat_cvt_f16<<<(nx + 255) / 256, 256, 0, stream>>>(x, xh, nx);
    gat_cvt_f16<<<(nw1 + 255) / 256, 256, 0, stream>>>(W1, W1h, nw1);

    gat_gemm_wmma<<<dim3(HIDTOT / BN, (N_NODES + BM - 1) / BM), 128, 0, stream>>>(
        xh, W1h, h1, N_NODES, HIDTOT, IN_DIM);

    // zero accumulators (after GEMM1: agg1 region overlays xh/W1h staging)
    (void)hipMemsetAsync(ws + 82560000, 0, 640000, stream);    // emax1 + den1
    (void)hipMemsetAsync(ws + 40960000, 0, 40960000, stream);  // agg1
    (void)hipMemsetAsync(ws + 88800000, 0, 160000, stream);    // emax2 + den2
    (void)hipMemsetAsync(ws + 90320000, 0, 5120000, stream);   // agg2

    gat_alpha_h4<<<N_NODES, 128, 0, stream>>>(h1, a_src1, a_dst1, as1, ad1);

    int th1 = E_TOT * HEADS;
    gat_edge_max<<<(th1 + 255) / 256, 256, 0, stream>>>(ei, HEADS, as1, ad1, emax1);
    gat_edge_exp<<<(th1 + 255) / 256, 256, 0, stream>>>(ei, HEADS, as1, ad1, emax1, eexp1, den1);
    gat_edge_agg<<<(E_TOT * 32 + 255) / 256, 256, 0, stream>>>(ei, HEADS, HIDTOT, h1, eexp1, den1, agg1);

    // ---- layer 2 ----
    int tot1 = N_NODES * HIDTOT, nw2 = HIDTOT * OUT_DIM;
    gat_relu_bias_f16<<<(tot1 + 255) / 256, 256, 0, stream>>>(agg1, b1, ah2, tot1, HIDTOT);
    gat_cvt_f16<<<(nw2 + 255) / 256, 256, 0, stream>>>(W2, W2h, nw2);

    gat_gemm_wmma<<<dim3(OUT_DIM / BN, (N_NODES + BM - 1) / BM), 128, 0, stream>>>(
        ah2, W2h, h2, N_NODES, OUT_DIM, HIDTOT);

    gat_alpha_h1<<<N_NODES, 64, 0, stream>>>(h2, a_src2, a_dst2, as2, ad2);

    gat_edge_max<<<(E_TOT + 255) / 256, 256, 0, stream>>>(ei, 1, as2, ad2, emax2);
    gat_edge_exp<<<(E_TOT + 255) / 256, 256, 0, stream>>>(ei, 1, as2, ad2, emax2, eexp2, den2);
    gat_edge_agg<<<(E_TOT * 32 + 255) / 256, 256, 0, stream>>>(ei, 1, OUT_DIM, h2, eexp2, den2, agg2);

    int tot2 = N_NODES * OUT_DIM;
    gat_bias_out<<<(tot2 + 255) / 256, 256, 0, stream>>>(agg2, b2, out, tot2, OUT_DIM);
}
